// CCAA_15298673508966
// MI455X (gfx1250) — matrix operand order repaired and checked by reading, not compiled, not run on hardware
//
#include <hip/hip_runtime.h>

// ===================== CDNA5 (gfx1250, wave32) bf16 WMMA helpers =====================
typedef __bf16 bf16;
typedef __attribute__((ext_vector_type(16))) __bf16 v16bf;
typedef __attribute__((ext_vector_type(8)))  __bf16 v8bf;
typedef __attribute__((ext_vector_type(8)))  float  v8f;

__device__ __forceinline__ bf16 f2bf(float f) {
  union { float f; unsigned u; } a; a.f = f;
  unsigned r = (a.u + 0x7FFFu + ((a.u >> 16) & 1u)) >> 16;   // round-to-nearest-even
  union { unsigned short s; bf16 b; } o; o.s = (unsigned short)r;
  return o.b;
}

__device__ __forceinline__ v8f vzero8() {
  v8f r;
#pragma unroll
  for (int i = 0; i < 8; ++i) r[i] = 0.f;
  return r;
}

// ---------------- Async global->LDS DMA (gfx1250 VGLOBAL async ops) ----------------
// VDST = VGPR with LDS byte offset (low 32 bits of a generic LDS pointer),
// VADDR = 64-bit global address. Tracked by ASYNCcnt -> s_wait_asynccnt.
__device__ __forceinline__ void async_g2l_b32(const void* g, const void* lds) {
#if defined(__gfx1250__)
  unsigned loff = (unsigned)(unsigned long long)lds;
  asm volatile("global_load_async_to_lds_b32 %0, %1, off"
               :: "v"(loff), "v"(g) : "memory");
#else
  *(float*)lds = *(const float*)g;
#endif
}
__device__ __forceinline__ void async_g2l_b128(const void* g, const void* lds) {
#if defined(__gfx1250__)
  unsigned loff = (unsigned)(unsigned long long)lds;
  asm volatile("global_load_async_to_lds_b128 %0, %1, off"
               :: "v"(loff), "v"(g) : "memory");
#else
  *(v8bf*)lds = *(const v8bf*)g;
#endif
}
__device__ __forceinline__ void wait_async0() {
#if defined(__gfx1250__)
  asm volatile("s_wait_asynccnt 0" ::: "memory");
#endif
}

// A-matrix 16x32 bf16 fragment. Lane l: m = l&15, khalf = l>>4.
// Lane holds K in two contiguous 8-element runs:
//   k in [khalf*8, khalf*8+8) and [16+khalf*8, 16+khalf*8+8)
__device__ __forceinline__ v16bf make_afrag(const bf16* rowbase, int khalf) {
  v8bf lo = *(const v8bf*)(rowbase + khalf * 8);
  v8bf hi = *(const v8bf*)(rowbase + 16 + khalf * 8);
  v16bf r;
#pragma unroll
  for (int e = 0; e < 8; ++e) { r[e] = lo[e]; r[e + 8] = hi[e]; }
  return r;
}

// B-matrix 32x16 bf16 fragment from a K-transposed LDS tile (smBt[n][k]).
// Lane l: n = l&15, khalf = l>>4; holds k in [khalf*8, khalf*8+8) and [16+khalf*8, 16+khalf*8+8).
__device__ __forceinline__ v16bf make_bfrag(const bf16* colbase, int khalf) {
  v8bf lo = *(const v8bf*)(colbase + khalf * 8);
  v8bf hi = *(const v8bf*)(colbase + 16 + khalf * 8);
  v16bf r;
#pragma unroll
  for (int e = 0; e < 8; ++e) { r[e] = lo[e]; r[e + 8] = hi[e]; }
  return r;
}

__device__ __forceinline__ v8f wmma_bf16(v16bf a, v16bf b, v8f c) {
  // (neg_a, A, neg_b, B, c_mod, C, reuse_a, reuse_b) -> v_wmma_f32_16x16x32_bf16
  return __builtin_amdgcn_wmma_f32_16x16x32_bf16(false, a, false, b, (short)0, c, false, false);
}

// ===================== Problem constants =====================
#define TC   64
#define TH   256
#define TW   256

// ===================== Small fp32 -> bf16 conversion (weights) =====================
__global__ void cvt_bf16(const float* __restrict__ in, bf16* __restrict__ out, int n) {
  const int i = blockIdx.x * 256 + threadIdx.x;
  if (i < n) out[i] = f2bf(in[i]);
}

// ===================== Dense conv as implicit GEMM (5x5 / dil-7x7 / 1x1) ============
// Workgroup: 128 threads (4 waves). Tile: 64 pixels x 64 oc.
// If wbf != null (1x1 projection), the bf16 [oc][ic] weight tile is DMA'd
// straight into LDS with global_load_async_to_lds_b128 (layouts match).
__global__ __launch_bounds__(128) void conv_gemm_wmma(
    const float* __restrict__ src, const float* __restrict__ wgt,
    const bf16* __restrict__ wbf,
    const float* __restrict__ bias, const float* __restrict__ res1,
    const float* __restrict__ res2, float* __restrict__ dst,
    int ks, int dil, int pad)
{
  const int b  = blockIdx.z;
  const int h  = blockIdx.y;
  const int w0 = blockIdx.x * 64;
  const int tid = threadIdx.x, lane = tid & 31, wv = tid >> 5;
  const int halfm = lane >> 4, n15 = lane & 15;

  __shared__ __align__(16) bf16 smA[64 * 72];   // A[p][ic], ldk=72 (144B rows)
  __shared__ __align__(16) bf16 smBt[64 * 72];  // B^T[oc][ic]

  v8f acc[4];
#pragma unroll
  for (int i = 0; i < 4; ++i) acc[i] = vzero8();

  const int taps = ks * ks;
  for (int t = 0; t < taps; ++t) {
    const int kh = t / ks, kw = t % ks;
    const int hy = h + kh * dil - pad;
    const int wxb = w0 + kw * dil - pad;
    const bool hok = (unsigned)hy < (unsigned)TH;
    __syncthreads();
    // Stage A: x[b, ic, hy, wxb+p]  (zero pad OOB), fp32 -> bf16
    for (int idx = tid; idx < 4096; idx += 128) {
      const int p = idx & 63, c = idx >> 6;
      const int wx = wxb + p;
      float v = 0.f;
      if (hok && (unsigned)wx < (unsigned)TW)
        v = src[(((b << 6) + c) << 16) + (hy << 8) + wx];
      smA[p * 72 + c] = f2bf(v);
    }
    // Stage B^T
    if (wbf) {
#if defined(__gfx1250__)
      // bf16 weights, [oc][ic] rows contiguous both sides: async DMA, 16B/lane
      for (int chunk = tid; chunk < 512; chunk += 128) {
        const int oc = chunk >> 3, icc = (chunk & 7) << 3;
        async_g2l_b128(wbf + (oc << 6) + icc, &smBt[oc * 72 + icc]);
      }
#else
      for (int idx = tid; idx < 4096; idx += 128) {
        const int ic = idx & 63, oc = idx >> 6;
        smBt[oc * 72 + ic] = wbf[(oc << 6) + ic];
      }
#endif
    } else {
      for (int idx = tid; idx < 4096; idx += 128) {
        const int ic = idx & 63, oc = idx >> 6;
        smBt[oc * 72 + ic] = f2bf(wgt[((oc << 6) + ic) * taps + t]);
      }
    }
    // Prefetch next tap's input row (global_prefetch_b8 on gfx1250)
    if (t + 1 < taps) {
      const int kh2 = (t + 1) / ks, kw2 = (t + 1) % ks;
      const int hy2 = h + kh2 * dil - pad;
      const int wx2 = w0 + kw2 * dil - pad + (tid & 63);
      if ((unsigned)hy2 < (unsigned)TH && (unsigned)wx2 < (unsigned)TW)
        __builtin_prefetch(&src[(((b << 6) + (tid >> 6)) << 16) + (hy2 << 8) + wx2], 0, 1);
    }
    if (wbf) wait_async0();          // async copies must land before the barrier
    __syncthreads();
#pragma unroll
    for (int kb = 0; kb < 2; ++kb) {
      const v16bf af = make_afrag(&smA[(wv * 16 + n15) * 72 + kb * 32], halfm);
#pragma unroll
      for (int nt = 0; nt < 4; ++nt) {
        const v16bf bfg = make_bfrag(&smBt[(nt * 16 + n15) * 72 + kb * 32], halfm);
        acc[nt] = wmma_bf16(af, bfg, acc[nt]);
      }
    }
  }

  // Epilogue: D layout -> row m = halfm*8+e (pixel), col n = oc
#pragma unroll
  for (int nt = 0; nt < 4; ++nt) {
    const int oc = nt * 16 + n15;
    const float bia = bias[oc];
#pragma unroll
    for (int e = 0; e < 8; ++e) {
      const int p = wv * 16 + halfm * 8 + e;
      const int off = (((b << 6) + oc) << 16) + (h << 8) + (w0 + p);
      float v = acc[nt][e] + bia;
      if (res1) v += res1[off];
      if (res2) v += res2[off];
      dst[off] = v;
    }
  }
}

// ===================== Depthwise 9x9 conv + residual =====================
// One block per output row (b,c,h). The 9 input rows (with w halo) are contiguous
// in both global and LDS -> staged with global_load_async_to_lds_b32; halo/padding
// zero-filled with plain LDS stores; s_wait_asynccnt 0 before the barrier.
__global__ __launch_bounds__(256) void dwconv9(
    const float* __restrict__ src, const float* __restrict__ wdw,
    const float* __restrict__ bias, const float* __restrict__ res,
    float* __restrict__ dst)
{
  const int bc = blockIdx.x >> 8;        // b*64 + c
  const int h  = blockIdx.x & 255;
  const int c  = bc & 63;
  const int t  = threadIdx.x;
  const float* plane = src + (bc << 16);

  __shared__ __align__(16) float smRow[9 * 264];   // rows h-4..h+4, cols -4..259

  for (int idx = t; idx < 9 * 264; idx += 256) {
    const int r = idx / 264, col = idx - r * 264;
    const int hy = h + r - 4, wx = col - 4;
    if ((unsigned)hy < (unsigned)TH && (unsigned)wx < (unsigned)TW) {
      async_g2l_b32(plane + (hy << 8) + wx, &smRow[idx]);
    } else {
      smRow[idx] = 0.f;                            // zero padding via ds_store
    }
  }
  wait_async0();
  __syncthreads();

  const float* wp = wdw + c * 81;
  float acc = bias[c];
#pragma unroll
  for (int kh = 0; kh < 9; ++kh)
#pragma unroll
    for (int kw = 0; kw < 9; ++kw)
      acc += smRow[kh * 264 + t + kw] * wp[kh * 9 + kw];

  const int gid = (blockIdx.x << 8) + t;
  dst[gid] = acc + res[gid];
}

// ===================== Per-row inverse L2 norms for split_h / split_w ===============
__global__ __launch_bounds__(256) void rownorm(
    const float* __restrict__ src, float* __restrict__ invn, int is, int js)
{
  const int rid = blockIdx.x;
  const int i = rid & 255, head = (rid >> 8) & 7, b = rid >> 11;
  const float* base = src + ((((b << 3) + head) << 3) << 16);
  const int t = threadIdx.x;
  float s = 0.f;
  for (int d = t; d < 2048; d += 256) {
    const int c = d & 7, j = d >> 3;
    const float v = base[(c << 16) + i * is + j * js];
    s += v * v;
  }
  __shared__ float red[256];
  red[t] = s; __syncthreads();
  for (int k = 128; k > 0; k >>= 1) { if (t < k) red[t] += red[t + k]; __syncthreads(); }
  if (t == 0) {
    const float n = sqrtf(red[0]);
    invn[rid] = 1.f / fmaxf(n, 1e-12f);
  }
}

// ===================== Axis attention (flash-style, full 256-wide rows) =============
__global__ __launch_bounds__(128) void attn_wmma(
    const float* __restrict__ Qsrc, const float* __restrict__ KVsrc,
    const float* __restrict__ invnq, const float* __restrict__ invnk,
    float* __restrict__ dst, int is, int js)
{
  const int b = blockIdx.z, head = blockIdx.y, row0 = blockIdx.x * 64;
  const int tid = threadIdx.x, lane = tid & 31, wv = tid >> 5;
  const int halfm = lane >> 4, n15 = lane & 15;
  const int cb = (((b << 3) + head) << 19);        // (b*64+head*8)*65536
  const int nb = (((b << 3) + head) << 8);         // norm-array row base

  __shared__ __align__(16) char smem[60672];
  bf16*  smQ  = (bf16*)smem;                        // 64*40   (phase 1)
  bf16*  smKt = smQ + 64 * 40;                      // 256*40  (phase 1)
  bf16*  smVt = (bf16*)smem;                        // 32*264  (phase 2, aliases Q/K)
  bf16*  smP  = (bf16*)(smem + 25600);              // 64*264  softmax probs
  float* smIQ = (float*)(smem + 59392);             // 64
  float* smIK = smIQ + 64;                          // 256

  if (tid < 64) smIQ[tid] = invnq[nb + row0 + tid];
  for (int g = tid; g < 256; g += 128) smIK[g] = invnk[nb + g];

  v8f acc[16];
#pragma unroll
  for (int i = 0; i < 16; ++i) acc[i] = vzero8();

  // ---- Phase 1: S accumulation over d = (j, c), K-chunks of 32 ----
  for (int k0 = 0; k0 < 2048; k0 += 32) {
    __syncthreads();
    for (int idx = tid; idx < 2048; idx += 128) {   // Q tile 64x32
      const int kk = idx & 31, r = idx >> 5;
      const int d = k0 + kk, c = d & 7, j = d >> 3;
      smQ[r * 40 + kk] = f2bf(Qsrc[cb + (c << 16) + (row0 + r) * is + j * js]);
    }
    for (int idx = tid; idx < 8192; idx += 128) {   // K tile 256x32 (transposed store)
      const int kk = idx & 31, g = idx >> 5;
      const int d = k0 + kk, c = d & 7, j = d >> 3;
      smKt[g * 40 + kk] = f2bf(KVsrc[cb + (c << 16) + g * is + j * js]);
    }
    __syncthreads();
    const v16bf af = make_afrag(&smQ[(wv * 16 + n15) * 40], halfm);
#pragma unroll
    for (int nt = 0; nt < 16; ++nt) {
      const v16bf bfg = make_bfrag(&smKt[(nt * 16 + n15) * 40], halfm);
      acc[nt] = wmma_bf16(af, bfg, acc[nt]);
    }
  }

  // ---- Softmax: scale by invnq[row]*invnk[col], rowwise max/sum via shfl_xor ----
  __syncthreads();
#pragma unroll
  for (int e = 0; e < 8; ++e) {
    const int r = wv * 16 + halfm * 8 + e;
    const float iq = smIQ[r];
    float sv[16];
    float mx = -3.4e38f;
#pragma unroll
    for (int nt = 0; nt < 16; ++nt) {
      const float s = acc[nt][e] * iq * smIK[nt * 16 + n15];
      sv[nt] = s;
      mx = fmaxf(mx, s);
    }
    for (int m = 8; m >= 1; m >>= 1) mx = fmaxf(mx, __shfl_xor(mx, m, 32));
    float sum = 0.f;
#pragma unroll
    for (int nt = 0; nt < 16; ++nt) { const float ev = __expf(sv[nt] - mx); sv[nt] = ev; sum += ev; }
    for (int m = 8; m >= 1; m >>= 1) sum += __shfl_xor(sum, m, 32);
    const float inv = 1.f / sum;
#pragma unroll
    for (int nt = 0; nt < 16; ++nt) smP[r * 264 + nt * 16 + n15] = f2bf(sv[nt] * inv);
  }

  // ---- Phase 2: O = P * V (+ normalized-Q residual), d-chunks of 32 ----
  for (int d0 = 0; d0 < 2048; d0 += 32) {
    __syncthreads();
    for (int idx = tid; idx < 32 * 256; idx += 128) {   // V^T tile [dd][g]
      const int g = idx & 255, dd = idx >> 8;
      const int d = d0 + dd, c = d & 7, j = d >> 3;
      smVt[dd * 264 + g] = f2bf(KVsrc[cb + (c << 16) + g * is + j * js]);
    }
    __syncthreads();
    v8f oa[2];
#pragma unroll
    for (int i = 0; i < 2; ++i) oa[i] = vzero8();
#pragma unroll
    for (int kb = 0; kb < 8; ++kb) {
      const v16bf af = make_afrag(&smP[(wv * 16 + n15) * 264 + kb * 32], halfm);
#pragma unroll
      for (int nt = 0; nt < 2; ++nt) {
        const v16bf bfg = make_bfrag(&smVt[(nt * 16 + n15) * 264 + kb * 32], halfm);
        oa[nt] = wmma_bf16(af, bfg, oa[nt]);
      }
    }
#pragma unroll
    for (int nt = 0; nt < 2; ++nt) {
      const int d = d0 + nt * 16 + n15, c = d & 7, j = d >> 3;
#pragma unroll
      for (int e = 0; e < 8; ++e) {
        const int r = wv * 16 + halfm * 8 + e;
        const int off = cb + (c << 16) + (row0 + r) * is + j * js;
        dst[off] = oa[nt][e] + Qsrc[off] * smIQ[r];
      }
    }
  }
}

// ===================== Host orchestration =====================
extern "C" void kernel_launch(void* const* d_in, const int* in_sizes, int n_in,
                              void* d_out, int out_size, void* d_ws, size_t ws_size,
                              hipStream_t stream) {
  (void)in_sizes; (void)n_in; (void)out_size; (void)ws_size;
  const float* x1     = (const float*)d_in[0];
  const float* x2     = (const float*)d_in[1];
  const float* w_pt   = (const float*)d_in[2];
  const float* b_pt   = (const float*)d_in[3];
  const float* w_d    = (const float*)d_in[4];
  const float* b_d    = (const float*)d_in[5];
  const float* w_dw   = (const float*)d_in[6];
  const float* b_dw   = (const float*)d_in[7];
  const float* w_proj = (const float*)d_in[8];
  const float* b_proj = (const float*)d_in[9];
  float* out = (float*)d_out;

  const size_t N = (size_t)4 * 64 * 256 * 256;    // one tensor = 16.78M floats
  float* s0 = (float*)d_ws;      // x11 -> t3 (merged out3)
  float* s1 = s0 + N;            // x12 -> t4 (merged out4)
  float* s2 = s1 + N;            // x1c -> x21 -> x2c
  float* s3 = s2 + N;            // out1
  float* s4 = s3 + N;            // x22 -> out2
  float* nrm = s4 + N;           // 4 x 8192 inverse row norms
  float* nh1 = nrm, * nh2 = nrm + 8192, * nw1 = nrm + 16384, * nw2 = nrm + 24576;
  bf16* wbf = (bf16*)(nrm + 32768);               // bf16 copy of w_proj (4096)

  const dim3 cg(4, 256, 4), cb(128);

  // bf16 projection weights (enables async B-tile DMA in the 1x1 GEMMs)
  cvt_bf16<<<16, 256, 0, stream>>>(w_proj, wbf, 4096);

  // conv chains (implicit-GEMM bf16 WMMA)
  conv_gemm_wmma<<<cg, cb, 0, stream>>>(x1, w_pt, nullptr, b_pt, x1, nullptr, s0, 5, 1, 2);  // x11
  conv_gemm_wmma<<<cg, cb, 0, stream>>>(s0, w_d,  nullptr, b_d,  s0, nullptr, s1, 7, 2, 6);  // x12
  dwconv9      <<<65536, 256, 0, stream>>>(s1, w_dw, b_dw, s1, s2);                          // x1c
  conv_gemm_wmma<<<cg, cb, 0, stream>>>(s2, w_proj, wbf, b_proj, nullptr, nullptr, s3, 1, 1, 0); // out1
  conv_gemm_wmma<<<cg, cb, 0, stream>>>(x2, w_pt, nullptr, b_pt, x2, nullptr, s2, 5, 1, 2);  // x21
  conv_gemm_wmma<<<cg, cb, 0, stream>>>(s2, w_d,  nullptr, b_d,  s2, nullptr, s4, 7, 2, 6);  // x22
  dwconv9      <<<65536, 256, 0, stream>>>(s4, w_dw, b_dw, s1, s2);                          // x2c (+x12 quirk)
  conv_gemm_wmma<<<cg, cb, 0, stream>>>(s2, w_proj, wbf, b_proj, nullptr, nullptr, s4, 1, 1, 0); // out2

  // inverse L2 norms of the four split views
  rownorm<<<8192, 256, 0, stream>>>(s3, nh1, 256, 1);   // split_h(out1)
  rownorm<<<8192, 256, 0, stream>>>(s4, nh2, 256, 1);   // split_h(out2)
  rownorm<<<8192, 256, 0, stream>>>(s3, nw1, 1, 256);   // split_w(out1)
  rownorm<<<8192, 256, 0, stream>>>(s4, nw2, 1, 256);   // split_w(out2)

  // axis attentions, writing merged NCHW results directly
  const dim3 ag(4, 8, 4);
  attn_wmma<<<ag, cb, 0, stream>>>(s4, s3, nh2, nh1, s0, 256, 1);   // out3 -> t3 (H-axis)
  attn_wmma<<<ag, cb, 0, stream>>>(s3, s4, nw1, nw2, s1, 1, 256);   // out4 -> t4 (W-axis)

  // final projections + residuals: out = proj(t3)+x1, then += proj(t4)+x2
  conv_gemm_wmma<<<cg, cb, 0, stream>>>(s0, w_proj, wbf, b_proj, x1, nullptr, out, 1, 1, 0);
  conv_gemm_wmma<<<cg, cb, 0, stream>>>(s1, w_proj, wbf, b_proj, x2, out, out, 1, 1, 0);
}
